// LorentzCGGNN_36163624633141
// MI455X (gfx1250) — matrix operand order repaired and checked by reading, not compiled, not run on hardware
//
#include <hip/hip_runtime.h>

#define N_NODES_C 1664
#define E_C       41600
#define HHC       72

typedef __bf16 bf16;
typedef __attribute__((ext_vector_type(16))) bf16  v16bf;
typedef __attribute__((ext_vector_type(8)))  float v8f;

// Cl(1,3) blade tables (blade order: grade-sorted combinations as in reference)
__constant__ int   c_BITS[16] = {0,1,2,4,8,3,5,9,6,10,12,7,11,13,14,15};
__constant__ int   c_INVB[16] = {0,1,2,5,3,6,8,11,4,7,9,12,10,13,14,15};
__constant__ int   c_GR[16]   = {0,1,1,1,1,2,2,2,2,2,2,3,3,3,3,4};
__constant__ float c_BB[16]   = {1.f,1.f,-1.f,-1.f,-1.f,-1.f,-1.f,-1.f,
                                 1.f,1.f, 1.f, 1.f, 1.f, 1.f,-1.f,-1.f};

__device__ __forceinline__ v8f wmma_bf16f32(v16bf a, v16bf b, v8f c) {
  return __builtin_amdgcn_wmma_f32_16x16x32_bf16(false, a, false, b, (short)0, c, false, false);
}

__device__ __forceinline__ float cayley_sign(int bi, int bk) {
  int s = 0, a = bi >> 1;
  while (a) { s += __popc(a & bk); a >>= 1; }
  float sgn = (s & 1) ? -1.f : 1.f;
  if (__popc(bi & bk & 0xE) & 1) sgn = -sgn;   // METRIC = (+,-,-,-)
  return sgn;
}

// ---------------- weight packing ----------------
// Packed B layout: [(kchunk*ntiles + ntile)*32 + lane]*16 + t  (bf16),
// element (lane,t) holds B[k = kchunk*32 + (lane>>4)*16 + t, n = ntile*16 + (lane&15)]
__global__ void pack_w_kernel(const float* __restrict__ W, int K, int N, int ntiles,
                              bf16* __restrict__ out) {
  int total = ((K + 31) / 32) * ntiles * 512;
  int p = blockIdx.x * blockDim.x + threadIdx.x;
  if (p >= total) return;
  int t    = p & 15;
  int lane = (p >> 4) & 31;
  int nt   = (p >> 9) % ntiles;
  int kc   = p / (512 * ntiles);
  int k = kc * 32 + (lane >> 4) * 16 + t;
  int n = nt * 16 + (lane & 15);
  float v = (k < K && n < N) ? W[(size_t)k * N + n] : 0.f;
  out[p] = (bf16)v;
}

// Densify Cayley-expanded geometric-product weights into the packed-B format.
// K = NCH*256 (flat n*256 + i*16 + k2), Ncols = 128 (m*16 + j)
__global__ void pack_gp_kernel(const float* __restrict__ w_gp, int NCH,
                               bf16* __restrict__ out) {
  int K = NCH * 256;
  int total = (K / 32) * 8 * 512;
  int p = blockIdx.x * blockDim.x + threadIdx.x;
  if (p >= total) return;
  int t    = p & 15;
  int lane = (p >> 4) & 31;
  int nt   = (p >> 9) & 7;
  int kc   = p >> 12;
  int k   = kc * 32 + (lane >> 4) * 16 + t;
  int col = nt * 16 + (lane & 15);
  int nch = k >> 8, i = (k >> 4) & 15, k2 = k & 15;
  int m = col >> 4, j = col & 15;
  int bi = c_BITS[i], bk = c_BITS[k2];
  float v = 0.f;
  if (c_INVB[bi ^ bk] == j) {
    v = cayley_sign(bi, bk) *
        w_gp[(((size_t)(m * NCH + nch) * 5 + c_GR[i]) * 5 + c_GR[j]) * 5 + c_GR[k2]];
  }
  out[p] = (bf16)v;
}

// ---------------- fused FCSGP kernel (the heavy bilinear) ----------------
// Phase 3 processes one channel n per "window": the A window is the outer
// product x[:,n,:] (x) r[:,n,:], built once per block into LDS as packed bf16
// pairs; all 8 waves then stream it with ds_load_b128.
template <int NCH>
__global__ __launch_bounds__(256)
void gp_fcsgp_kernel(const float* __restrict__ xnode,   // [N,128]
                     const float* __restrict__ aux0,    // edge: edge_attr [E,48]; node: x_red [N,128]
                     const float* __restrict__ aux1,    // node: node_attr_x [N,16]
                     const int* __restrict__ ei, const int* __restrict__ ej,
                     const float* __restrict__ w_right, // [NCH,NCH,5]
                     const float* __restrict__ w_left,  // [8,NCH,5]
                     const float* __restrict__ b_left,  // [8]
                     const bf16* __restrict__ Bpack,    // packed gp weights
                     float* __restrict__ out,           // [R,128]
                     int edge_mode) {
  __shared__ float    sx[16][NCH * 16 + 4];
  __shared__ float    sr[16][NCH * 16 + 4];
  __shared__ unsigned sa[16][132];           // one window: 128 u32 (256 bf16) per row
  const int tid  = threadIdx.x;
  const int row0 = blockIdx.x * 16;

  // Phase 1: gather/concat input multivector rows
  for (int p = tid; p < 16 * NCH * 16; p += 256) {
    int r  = p / (NCH * 16);
    int ci = p % (NCH * 16);
    int c = ci >> 4, blade = ci & 15;
    int gr = row0 + r;
    float v;
    if (edge_mode) {
      int vi = ei[gr], vj = ej[gr];
      if (c < 8)       v = xnode[(size_t)vi * 128 + c * 16 + blade];
      else if (c < 16) v = xnode[(size_t)vj * 128 + (c - 8) * 16 + blade];
      else if (c < 24) v = xnode[(size_t)vi * 128 + (c - 16) * 16 + blade] -
                           xnode[(size_t)vj * 128 + (c - 16) * 16 + blade];
      else             v = aux0[(size_t)gr * 48 + (c - 24) * 16 + blade];
    } else {
      if (c < 8)       v = xnode[(size_t)gr * 128 + c * 16 + blade];
      else if (c < 16) v = aux0[(size_t)gr * 128 + (c - 8) * 16 + blade];
      else             v = aux1[(size_t)gr * 16 + blade];
    }
    sx[r][ci] = v;
  }
  __syncthreads();

  // Phase 2: r = mv_linear(m_in, w_right)  (per-blade NCHxNCH matmul)
  for (int p = tid; p < 16 * NCH * 16; p += 256) {
    int r  = p / (NCH * 16);
    int ci = p % (NCH * 16);
    int m = ci >> 4, blade = ci & 15;
    int g = c_GR[blade];
    float acc = 0.f;
    for (int n = 0; n < NCH; ++n)
      acc += sx[r][n * 16 + blade] * w_right[(size_t)(m * NCH + n) * 5 + g];
    sr[r][ci] = acc;
  }
  __syncthreads();

  // Phase 3: per-channel windows; cooperative A build + 8 WMMA steps each
  const int wave = tid >> 5, lane = tid & 31;
  const int mrow = lane & 15, half = lane >> 4;
  v8f acc = {0.f, 0.f, 0.f, 0.f, 0.f, 0.f, 0.f, 0.f};
  const v16bf* bp = (const v16bf*)Bpack + wave * 32 + lane;  // step 256 per kstep
  for (int w = 0; w < NCH; ++w) {
    // build packed A window: u32 j = i*8 + kpair  (covers k = 2*kpair, 2*kpair+1)
    for (int p = tid; p < 2048; p += 256) {
      int r = p >> 7, j = p & 127;
      int i = j >> 3, kp = j & 7;
      float xv = sx[r][w * 16 + i];
      float r0 = sr[r][w * 16 + 2 * kp];
      float r1 = sr[r][w * 16 + 2 * kp + 1];
      union { bf16 h[2]; unsigned u; } pk;
      pk.h[0] = (bf16)(xv * r0); pk.h[1] = (bf16)(xv * r1);
      sa[r][j] = pk.u;
    }
    __syncthreads();
#pragma unroll
    for (int s = 0; s < 8; ++s) {
      const int kc = w * 8 + s;
      union { v16bf v; unsigned u[8]; } a;
#pragma unroll
      for (int q = 0; q < 8; ++q) {
        int jj = s * 16 + 8 * half + q;
        a.u[q] = sa[mrow][jj];
      }
      v16bf b = bp[(size_t)kc * 256];
      acc = wmma_bf16f32(a.v, b, acc);
    }
    __syncthreads();
  }

  // Phase 4: epilogue — add mv_linear(m_in, w_left, b_left), scale 1/sqrt(2)
  const int ncol = lane & 15;
  const int col  = wave * 16 + ncol;
  const int mch  = col >> 4, blade = col & 15;
  const int g    = c_GR[blade];
#pragma unroll
  for (int v = 0; v < 8; ++v) {
    int r = v + 8 * half;
    float lv = (blade == 0) ? b_left[mch] : 0.f;
    for (int n = 0; n < NCH; ++n)
      lv += sx[r][n * 16 + blade] * w_left[(size_t)(mch * NCH + n) * 5 + g];
    out[(size_t)(row0 + r) * 128 + col] = (lv + acc[v]) * 0.70710678118654752f;
  }
}

// ---------------- generic bf16 WMMA GEMM: C = act(A @ W + b), K <= 256 ----------------
// A tile staged once in LDS as packed bf16 pairs; 8 waves cover the N tiles.
__global__ __launch_bounds__(256)
void gemm_wmma_kernel(const float* __restrict__ A, int lda,
                      const bf16* __restrict__ Bpack,
                      const float* __restrict__ bias,
                      float* __restrict__ C, int ldc,
                      int K, int N, int act) {
  __shared__ unsigned sA[16][132];   // u32 = packed bf16 pair; 132 pad -> conflict-free
  const int tid = threadIdx.x;
  const int mrow0 = blockIdx.x * 16;
  const int nkc = (K + 31) / 32;
  const int tot = 16 * nkc * 16;     // u32 elements to stage
  for (int p = tid; p < tot; p += 256) {
    int r = p / (nkc * 16), j = p % (nkc * 16);
    int k0 = 2 * j;
    float f0 = (k0 < K)     ? A[(size_t)(mrow0 + r) * lda + k0]     : 0.f;
    float f1 = (k0 + 1 < K) ? A[(size_t)(mrow0 + r) * lda + k0 + 1] : 0.f;
    union { bf16 h[2]; unsigned u; } pk;
    pk.h[0] = (bf16)f0; pk.h[1] = (bf16)f1;
    sA[r][j] = pk.u;
  }
  __syncthreads();
  const int wave = tid >> 5, lane = tid & 31;
  const int ntiles = (N + 15) >> 4;
  if (wave >= ntiles) return;
  const int mrow = lane & 15, half = lane >> 4;
  v8f acc = {0.f, 0.f, 0.f, 0.f, 0.f, 0.f, 0.f, 0.f};
  const v16bf* bp = (const v16bf*)Bpack + wave * 32 + lane;  // step ntiles*32 per kstep
  for (int kc = 0; kc < nkc; ++kc) {
    union { v16bf v; unsigned u[8]; } a;
#pragma unroll
    for (int q = 0; q < 8; ++q) {
      int jj = kc * 16 + 8 * half + q;
      a.u[q] = sA[mrow][jj];
    }
    __builtin_prefetch((const void*)(bp + (size_t)(kc + 1) * ntiles * 32), 0, 3);
    v16bf b = bp[(size_t)kc * ntiles * 32];
    acc = wmma_bf16f32(a.v, b, acc);
  }
  const int col = wave * 16 + (lane & 15);
  if (col < N) {
    float bv = bias ? bias[col] : 0.f;
#pragma unroll
    for (int v = 0; v < 8; ++v) {
      int r = mrow0 + v + 8 * half;
      float val = acc[v] + bv;
      if (act == 1) val = fmaxf(val, 0.f);
      C[(size_t)r * ldc + col] = val;
    }
  }
}

// ---------------- small elementwise / reduction kernels ----------------
__global__ void embed_kernel(const float* __restrict__ h_in, const float* __restrict__ x_in,
                             const float* __restrict__ Wh, const float* __restrict__ bh,
                             const float* __restrict__ Wx, const float* __restrict__ bx,
                             float* __restrict__ h0, float* __restrict__ x0) {
  int p = blockIdx.x * blockDim.x + threadIdx.x;
  const int NH = N_NODES_C * HHC;
  if (p < NH) {
    int v = p / HHC, c = p % HHC;
    h0[p] = h_in[v * 2] * Wh[c] + h_in[v * 2 + 1] * Wh[HHC + c] + bh[c];
  } else if (p < NH + N_NODES_C * 128) {
    int q = p - NH;
    int v = q >> 7, ci = q & 127, m = ci >> 4, i = ci & 15;
    x0[q] = Wx[m] * x_in[v * 16 + i] + ((i == 0) ? bx[m] : 0.f);
  }
}

__global__ void mvln_kernel(float* __restrict__ x, const float* __restrict__ a, int R) {
  int r = blockIdx.x * blockDim.x + threadIdx.x;
  if (r >= R) return;
  float* xr = x + (size_t)r * 128;
  float s = 0.f;
  for (int c = 0; c < 8; ++c) {
    float acc = 0.f;
    for (int i = 0; i < 16; ++i) { float v = xr[c * 16 + i]; acc += c_BB[i] * v * v; }
    s += sqrtf(fabsf(acc));
  }
  float norm = s * 0.125f + 1e-6f;
  for (int c = 0; c < 8; ++c) {
    float sc = a[c] / norm;
    for (int i = 0; i < 16; ++i) xr[c * 16 + i] *= sc;
  }
}

__device__ __forceinline__ void compute_inv(const float* xr, float* inv) {
  for (int c = 0; c < 8; ++c) {
    float q[5] = {0.f, 0.f, 0.f, 0.f, 0.f};
    for (int i = 0; i < 16; ++i) { float v = xr[c * 16 + i]; q[c_GR[i]] += c_BB[i] * v * v; }
    inv[c * 5] = xr[c * 16];
    for (int g = 1; g < 5; ++g) inv[c * 5 + g] = q[g];
  }
}

__global__ void mh_in_kernel(const float* __restrict__ m_x, const float* __restrict__ h,
                             const int* __restrict__ ei, const int* __restrict__ ej,
                             float* __restrict__ mh, int E) {
  int e = blockIdx.x * blockDim.x + threadIdx.x;
  if (e >= E) return;
  float inv[40];
  compute_inv(m_x + (size_t)e * 128, inv);
  float* o = mh + (size_t)e * 256;
  for (int t = 0; t < 40; ++t) o[t] = inv[t];
  const float* hi = h + (size_t)ei[e] * HHC;
  const float* hj = h + (size_t)ej[e] * HHC;
  for (int c = 0; c < HHC; ++c) {
    o[40 + c] = hi[c]; o[112 + c] = hj[c]; o[184 + c] = hi[c] - hj[c];
  }
}

__global__ void col_stats_kernel(const float* __restrict__ A, int M, int ld,
                                 float* __restrict__ stats, int N) {
  int col = blockIdx.x;
  __shared__ float s1[256], s2[256];
  float a1 = 0.f, a2 = 0.f;
  for (int r = threadIdx.x; r < M; r += 256) {
    float v = A[(size_t)r * ld + col]; a1 += v; a2 += v * v;
  }
  s1[threadIdx.x] = a1; s2[threadIdx.x] = a2; __syncthreads();
  for (int st = 128; st > 0; st >>= 1) {
    if (threadIdx.x < st) { s1[threadIdx.x] += s1[threadIdx.x + st]; s2[threadIdx.x] += s2[threadIdx.x + st]; }
    __syncthreads();
  }
  if (threadIdx.x == 0) {
    float mean = s1[0] / (float)M;
    stats[col] = mean;
    stats[N + col] = s2[0] / (float)M - mean * mean;
  }
}

__global__ void bn_relu_kernel(float* __restrict__ A, int M, int ld, int N,
                               const float* __restrict__ stats,
                               const float* __restrict__ g, const float* __restrict__ b) {
  int p = blockIdx.x * blockDim.x + threadIdx.x;
  if (p >= M * N) return;
  int r = p / N, c = p % N;
  float v = (A[(size_t)r * ld + c] - stats[c]) * rsqrtf(stats[N + c] + 1e-5f) * g[c] + b[c];
  A[(size_t)r * ld + c] = fmaxf(v, 0.f);
}

__global__ void gate_kernel(float* __restrict__ x, const float* __restrict__ w, int R) {
  int p = blockIdx.x * blockDim.x + threadIdx.x;
  if (p >= R * 128) return;
  int r = p >> 7, ci = p & 127, c = ci >> 4, i = ci & 15;
  float wv = w[(size_t)r * 40 + c * 5 + c_GR[i]];
  x[p] *= 1.f / (1.f + expf(-wv));
}

__global__ void segmean_kernel(const float* __restrict__ src, int ld, int width,
                               float* __restrict__ dst, int dld) {
  int p = blockIdx.x * blockDim.x + threadIdx.x;
  if (p >= N_NODES_C * width) return;
  int v = p / width, c = p % width;
  float s = 0.f;
  for (int t = 0; t < 25; ++t) s += src[(size_t)(v * 25 + t) * ld + c];
  dst[(size_t)v * dld + c] = s * 0.04f;
}

__global__ void inv_kernel(const float* __restrict__ x, float* __restrict__ inv, int R) {
  int r = blockIdx.x * blockDim.x + threadIdx.x;
  if (r >= R) return;
  compute_inv(x + (size_t)r * 128, inv + (size_t)r * 40);
}

__global__ void hu_in_kernel(const float* __restrict__ h, const float* __restrict__ h_red,
                             const float* __restrict__ u_inv, const float* __restrict__ nah,
                             float* __restrict__ o) {
  int v = blockIdx.x * blockDim.x + threadIdx.x;
  if (v >= N_NODES_C) return;
  float* p = o + (size_t)v * 186;
  for (int c = 0; c < HHC; ++c) { p[c] = h[(size_t)v * HHC + c]; p[HHC + c] = h_red[(size_t)v * HHC + c]; }
  for (int t = 0; t < 40; ++t) p[144 + t] = u_inv[(size_t)v * 40 + t];
  p[184] = nah[v * 2]; p[185] = nah[v * 2 + 1];
}

__global__ void final_concat_kernel(const float* __restrict__ h, const float* __restrict__ x,
                                    const float* __restrict__ mask, float* __restrict__ o) {
  int v = blockIdx.x * blockDim.x + threadIdx.x;
  if (v >= N_NODES_C) return;
  float inv[40];
  compute_inv(x + (size_t)v * 128, inv);
  float m = mask[v];
  for (int c = 0; c < HHC; ++c) o[(size_t)v * 112 + c] = h[(size_t)v * HHC + c] * m;
  for (int t = 0; t < 40; ++t)  o[(size_t)v * 112 + HHC + t] = inv[t] * m;
}

__global__ void pool_kernel(const float* __restrict__ o, float* __restrict__ pooled) {
  int p = blockIdx.x * blockDim.x + threadIdx.x;
  if (p >= 64 * 112) return;
  int g = p / 112, c = p % 112;
  float s = 0.f;
  for (int t = 0; t < 26; ++t) s += o[(size_t)(g * 26 + t) * 112 + c];
  pooled[p] = s * (1.f / 26.f);
}

// ---------------- host orchestration ----------------
static inline int cdiv_h(int a, int b) { return (a + b - 1) / b; }

extern "C" void kernel_launch(void* const* d_in, const int* in_sizes, int n_in,
                              void* d_out, int out_size, void* d_ws, size_t ws_size,
                              hipStream_t stream) {
  (void)in_sizes; (void)n_in; (void)out_size; (void)ws_size;
  const float* h_in  = (const float*)d_in[0];
  const float* x_in  = (const float*)d_in[1];
  const float* nah   = (const float*)d_in[2];
  const float* nax   = (const float*)d_in[3];
  const float* eax   = (const float*)d_in[4];
  const float* nmask = (const float*)d_in[5];
  const int*   edges = (const int*)d_in[6];
  // d_in[7] = n_nodes scalar (hardcoded 26)
  const float* dec_W1 = (const float*)d_in[8];
  const float* dec_W2 = (const float*)d_in[9];
  const float* dec_b1 = (const float*)d_in[10];
  const float* dec_b2 = (const float*)d_in[11];
  const float* emb_h_W = (const float*)d_in[12];
  const float* emb_h_b = (const float*)d_in[13];
  const float* emb_x_W = (const float*)d_in[14];
  const float* emb_x_b = (const float*)d_in[15];
  // params['layers'][l] pytree leaves (key-sorted), 29 per layer, base 16
  auto L = [&](int l, int k) -> const float* { return (const float*)d_in[16 + 29 * l + k]; };
  const int* ei = edges;
  const int* ej = edges + E_C;

  // workspace carve-up
  char* base = (char*)d_ws;
  size_t off = 0;
  auto take = [&](size_t bytes) -> void* {
    void* p = base + off; off = (off + bytes + 255) & ~(size_t)255; return p;
  };
  float* xA    = (float*)take((size_t)N_NODES_C * 128 * 4);
  float* xB    = (float*)take((size_t)N_NODES_C * 128 * 4);
  float* hA    = (float*)take((size_t)N_NODES_C * HHC * 4);
  float* hB    = (float*)take((size_t)N_NODES_C * HHC * 4);
  float* x_red = (float*)take((size_t)N_NODES_C * 128 * 4);
  float* h_red = (float*)take((size_t)N_NODES_C * HHC * 4);
  float* u_inv = (float*)take((size_t)N_NODES_C * 40 * 4);
  float* hu_in = (float*)take((size_t)N_NODES_C * 186 * 4);
  float* ndt   = (float*)take((size_t)N_NODES_C * HHC * 4);
  float* w_n   = (float*)take((size_t)N_NODES_C * 40 * 4);
  float* m_x   = (float*)take((size_t)E_C * 128 * 4);
  float* mh    = (float*)take((size_t)E_C * 256 * 4);
  float* t1    = (float*)take((size_t)E_C * HHC * 4);
  float* m_h   = (float*)take((size_t)E_C * HHC * 4);
  float* psi_t = (float*)take((size_t)E_C * HHC * 4);
  float* w_e   = (float*)take((size_t)E_C * 40 * 4);
  float* stats = (float*)take(2 * HHC * 4);
  float* cat   = (float*)take((size_t)N_NODES_C * 112 * 4);
  float* pooled= (float*)take(64 * 112 * 4);
  float* dec_t = (float*)take(64 * 64 * 4);
  bf16 *gpE[4], *gpN[4], *pW[4][8];
  for (int l = 0; l < 4; ++l) {
    gpE[l] = (bf16*)take((size_t)216 * 8 * 512 * 2);
    gpN[l] = (bf16*)take((size_t)136 * 8 * 512 * 2);
    pW[l][0] = (bf16*)take((size_t)8 * 5 * 512 * 2);  // phi_h_W1 [256,72]
    pW[l][1] = (bf16*)take((size_t)3 * 5 * 512 * 2);  // phi_h_W2 [72,72]
    pW[l][2] = (bf16*)take((size_t)3 * 5 * 512 * 2);  // psi_W1
    pW[l][3] = (bf16*)take((size_t)3 * 3 * 512 * 2);  // psi_W2 [72,40]
    pW[l][4] = (bf16*)take((size_t)6 * 5 * 512 * 2);  // theta_h_W1 [186,72]
    pW[l][5] = (bf16*)take((size_t)3 * 5 * 512 * 2);  // theta_h_W2
    pW[l][6] = (bf16*)take((size_t)3 * 5 * 512 * 2);  // chi_W1
    pW[l][7] = (bf16*)take((size_t)3 * 3 * 512 * 2);  // chi_W2
  }
  bf16* pdec1 = (bf16*)take((size_t)4 * 4 * 512 * 2);
  bf16* pdec2 = (bf16*)take((size_t)2 * 1 * 512 * 2);

  auto pack = [&](const float* W, int K, int N, bf16* out) {
    int ntiles = cdiv_h(N, 16);
    int total = cdiv_h(K, 32) * ntiles * 512;
    pack_w_kernel<<<cdiv_h(total, 256), 256, 0, stream>>>(W, K, N, ntiles, out);
  };
  auto gemm = [&](const float* A, int lda, const bf16* Bp, const float* bias,
                  float* C, int ldc, int M, int K, int N, int act) {
    gemm_wmma_kernel<<<M / 16, 256, 0, stream>>>(A, lda, Bp, bias, C, ldc, K, N, act);
  };

  // pack all weights (deterministic, every call)
  for (int l = 0; l < 4; ++l) {
    pack_gp_kernel<<<cdiv_h(216 * 4096, 256), 256, 0, stream>>>(L(l,10), 27, gpE[l]);
    pack_gp_kernel<<<cdiv_h(136 * 4096, 256), 256, 0, stream>>>(L(l,25), 17, gpN[l]);
    pack(L(l,4), 256, 72, pW[l][0]);
    pack(L(l,5),  72, 72, pW[l][1]);
    pack(L(l,14), 72, 72, pW[l][2]);
    pack(L(l,15), 72, 40, pW[l][3]);
    pack(L(l,18),186, 72, pW[l][4]);
    pack(L(l,19), 72, 72, pW[l][5]);
    pack(L(l,0),  72, 72, pW[l][6]);
    pack(L(l,1),  72, 40, pW[l][7]);
  }
  pack(dec_W1, 112, 64, pdec1);
  pack(dec_W2,  64,  2, pdec2);

  // embedding
  embed_kernel<<<cdiv_h(N_NODES_C * (HHC + 128), 256), 256, 0, stream>>>(
      h_in, x_in, emb_h_W, emb_h_b, emb_x_W, emb_x_b, hA, xA);

  float *hcur = hA, *hnxt = hB, *xcur = xA, *xnxt = xB;
  for (int l = 0; l < 4; ++l) {
    // phi_x: fused FCSGP over edges  -> m_x
    gp_fcsgp_kernel<27><<<E_C / 16, 256, 0, stream>>>(
        xcur, eax, (const float*)nullptr, ei, ej,
        L(l,12), L(l,11), L(l,9), gpE[l], m_x, 1);
    mvln_kernel<<<cdiv_h(E_C, 256), 256, 0, stream>>>(m_x, L(l,13), E_C);
    // phi_h MLP
    mh_in_kernel<<<cdiv_h(E_C, 256), 256, 0, stream>>>(m_x, hcur, ei, ej, mh, E_C);
    gemm(mh, 256, pW[l][0], nullptr, t1, HHC, E_C, 256, HHC, 0);
    col_stats_kernel<<<HHC, 256, 0, stream>>>(t1, E_C, HHC, stats, HHC);
    bn_relu_kernel<<<cdiv_h(E_C * HHC, 256), 256, 0, stream>>>(t1, E_C, HHC, HHC, stats, L(l,8), L(l,7));
    gemm(t1, HHC, pW[l][1], L(l,6), m_h, HHC, E_C, HHC, HHC, 1);
    // psi gating
    gemm(m_h, HHC, pW[l][2], L(l,16), psi_t, HHC, E_C, HHC, HHC, 1);
    gemm(psi_t, HHC, pW[l][3], L(l,17), w_e, 40, E_C, HHC, 40, 0);
    gate_kernel<<<cdiv_h(E_C * 128, 256), 256, 0, stream>>>(m_x, w_e, E_C);
    // aggregation (25 consecutive edges per source node)
    segmean_kernel<<<cdiv_h(N_NODES_C * 128, 256), 256, 0, stream>>>(m_x, 128, 128, x_red, 128);
    segmean_kernel<<<cdiv_h(N_NODES_C * HHC, 256), 256, 0, stream>>>(m_h, HHC, HHC, h_red, HHC);
    // theta_x: fused FCSGP over nodes -> x_u
    inv_kernel<<<cdiv_h(N_NODES_C, 256), 256, 0, stream>>>(xcur, u_inv, N_NODES_C);
    gp_fcsgp_kernel<17><<<N_NODES_C / 16, 256, 0, stream>>>(
        xcur, x_red, nax, (const int*)nullptr, (const int*)nullptr,
        L(l,27), L(l,26), L(l,24), gpN[l], xnxt, 0);
    mvln_kernel<<<cdiv_h(N_NODES_C, 256), 256, 0, stream>>>(xnxt, L(l,28), N_NODES_C);
    // theta_h MLP
    hu_in_kernel<<<cdiv_h(N_NODES_C, 256), 256, 0, stream>>>(hcur, h_red, u_inv, nah, hu_in);
    gemm(hu_in, 186, pW[l][4], L(l,20), ndt, HHC, N_NODES_C, 186, HHC, 0);
    col_stats_kernel<<<HHC, 256, 0, stream>>>(ndt, N_NODES_C, HHC, stats, HHC);
    bn_relu_kernel<<<cdiv_h(N_NODES_C * HHC, 256), 256, 0, stream>>>(ndt, N_NODES_C, HHC, HHC, stats, L(l,23), L(l,22));
    gemm(ndt, HHC, pW[l][5], L(l,21), hnxt, HHC, N_NODES_C, HHC, HHC, 0);
    // chi gating
    gemm(hnxt, HHC, pW[l][6], L(l,2), ndt, HHC, N_NODES_C, HHC, HHC, 1);
    gemm(ndt, HHC, pW[l][7], L(l,3), w_n, 40, N_NODES_C, HHC, 40, 0);
    gate_kernel<<<cdiv_h(N_NODES_C * 128, 256), 256, 0, stream>>>(xnxt, w_n, N_NODES_C);
    // advance
    float* t;
    t = hcur; hcur = hnxt; hnxt = t;
    t = xcur; xcur = xnxt; xnxt = t;
  }

  // readout
  final_concat_kernel<<<cdiv_h(N_NODES_C, 256), 256, 0, stream>>>(hcur, xcur, nmask, cat);
  pool_kernel<<<cdiv_h(64 * 112, 256), 256, 0, stream>>>(cat, pooled);
  gemm(pooled, 112, pdec1, dec_b1, dec_t, 64, 64, 112, 64, 1);
  gemm(dec_t, 64, pdec2, dec_b2, (float*)d_out, 2, 64, 64, 2, 0);
}
